// Multi_head_attention_67765993997023
// MI455X (gfx1250) — matrix operand order repaired and checked by reading, not compiled, not run on hardware
//
#include <hip/hip_runtime.h>

typedef __attribute__((ext_vector_type(16))) _Float16 v16h;
typedef __attribute__((ext_vector_type(8)))  _Float16 v8h;
typedef __attribute__((ext_vector_type(8)))  float    v8f;
typedef __attribute__((ext_vector_type(4)))  float    v4f;

constexpr int kS  = 512;   // sequence length
constexpr int kD  = 512;   // model width
constexpr int kH  = 8;     // heads
constexpr int kDh = 64;    // head dim

// ---------------- workspace layout (bytes) ----------------
constexpr size_t KB = 1024;
constexpr size_t OFF_XQ  = 0;          // f16 query          512KB
constexpr size_t OFF_XK  = 512*KB;     // f16 key            512KB
constexpr size_t OFF_XV  = 1024*KB;    // f16 value          512KB
constexpr size_t OFF_WQT = 1536*KB;    // f16 Wq^T           512KB
constexpr size_t OFF_WKT = 2048*KB;    // f16 Wk^T           512KB
constexpr size_t OFF_WVT = 2560*KB;    // f16 Wv^T           512KB
constexpr size_t OFF_WR  = 3072*KB;    // f16 Wr (row-major) 512KB
constexpr size_t OFF_QU  = 3584*KB;    // f16 [h][s][d] q+u  512KB
constexpr size_t OFF_QV  = 4096*KB;    // f16 [h][s][d] q+v  512KB
constexpr size_t OFF_KH  = 4608*KB;    // f16 [h][s][d] k    512KB
constexpr size_t OFF_VT  = 5120*KB;    // f16 [h][d][s] v^T  512KB
constexpr size_t OFF_P   = 5632*KB;    // f16 [h][q][e]      4MB
constexpr size_t OFF_SC  = 9728*KB;    // f32 [h][q][k]      8MB
constexpr size_t OFF_AT  = 17920*KB;   // f16 [h][q][k] attn 4MB
constexpr size_t OFF_CB  = 22016*KB;   // f32 [h][q] bias c  16KB
constexpr size_t OFF_OT  = 22032*KB;   // f32 [s][o] out tmp 1MB
// total ~22.5 MB

// ---------------- fragment loaders (wave32 CDNA5 WMMA layouts) ----------------
// A 16x32 f16: lane l (half=l>>4, m=l&15): elems0-15 = K[k0+half*16 .. +15];
//              row = m0+m.
__device__ inline v16h load_a_f16(const _Float16* __restrict__ base, int ld,
                                  int m0, int k0, int lane) {
  int h4 = lane >> 4, r = lane & 15;
  const _Float16* p = base + (size_t)(m0 + r) * ld + k0 + h4 * 16;
  v8h lo = *(const v8h*)p;
  v8h hi = *(const v8h*)(p + 8);
  return __builtin_shufflevector(lo, hi, 0,1,2,3,4,5,6,7,8,9,10,11,12,13,14,15);
}
// B 32x16 f16: lane l (half=l>>4, n=l&15): elems0-15 = K[k0+half*16 .. +15],
// column n. Source stored K-contiguous per column (i.e. [n][K], ld given).
__device__ inline v16h load_b_f16(const _Float16* __restrict__ base, int ld,
                                  int n0, int k0, int lane) {
  int h4 = lane >> 4, r = lane & 15;
  return *(const v16h*)(base + (size_t)(n0 + r) * ld + k0 + h4 * 16);
}

// ---------------- elementwise f32 -> f16 ----------------
__global__ void cvt_f32_to_f16(const float* __restrict__ src,
                               _Float16* __restrict__ dst, int n) {
  int i = blockIdx.x * blockDim.x + threadIdx.x;
  if (i < n) dst[i] = (_Float16)src[i];
}

// ---------------- transpose + cvt for projection weights ----------------
__global__ void transpose_cvt(const float* __restrict__ W, _Float16* __restrict__ Wt) {
  __shared__ float tile[32][33];
  int bx = blockIdx.x * 32, by = blockIdx.y * 32;
  int tx = threadIdx.x & 31, ty = threadIdx.x >> 5;  // 256 threads: 32 x 8
  #pragma unroll
  for (int j = 0; j < 32; j += 8)
    tile[ty + j][tx] = W[(size_t)(by + ty + j) * kD + bx + tx];
  __syncthreads();
  #pragma unroll
  for (int j = 0; j < 32; j += 8)
    Wt[(size_t)(bx + ty + j) * kD + by + tx] = (_Float16)tile[tx][ty + j];
}

// ---------------- fused QKV projection (WMMA) ----------------
// mode 0: Q -> outA=q+u, outB=q+v  ([h][s][d] f16)
// mode 1: K -> outA=k              ([h][s][d] f16)
// mode 2: V -> outA=v^T            ([h][d][s] f16)
__global__ void gemm_proj(const _Float16* __restrict__ X, const _Float16* __restrict__ Wt,
                          const float* __restrict__ bias,
                          const float* __restrict__ uvec, const float* __restrict__ vvec,
                          _Float16* __restrict__ outA, _Float16* __restrict__ outB,
                          int mode) {
  int wave = blockIdx.x * 8 + (threadIdx.x >> 5);
  int lane = threadIdx.x & 31;
  int m0 = (wave >> 5) * 16, n0 = (wave & 31) * 16;
  v8f acc = {};
  for (int k0 = 0; k0 < kD; k0 += 32) {
    v16h a = load_a_f16(X, kD, m0, k0, lane);
    v16h b = load_b_f16(Wt, kD, n0, k0, lane);
    acc = __builtin_amdgcn_wmma_f32_16x16x32_f16(false, a, false, b,
                                                 (short)0, acc, false, false);
  }
  int h4 = lane >> 4, r = lane & 15;
  int o = n0 + r;            // output feature
  int h = o >> 6, d = o & 63;
  float bo = bias[o];
  #pragma unroll
  for (int j = 0; j < 8; j++) {
    int srow = m0 + j + h4 * 8;
    float val = acc[j] + bo;
    if (mode == 0) {
      outA[((size_t)h * kS + srow) * kDh + d] = (_Float16)(val + uvec[o]);
      outB[((size_t)h * kS + srow) * kDh + d] = (_Float16)(val + vvec[o]);
    } else if (mode == 1) {
      outA[((size_t)h * kS + srow) * kDh + d] = (_Float16)val;
    } else {
      outA[((size_t)h * kDh + d) * kS + srow] = (_Float16)val;  // V transposed
    }
  }
}

// ---------------- 64-deep GEMM helper (K = Dh) ----------------
__device__ inline v8f gemm_k64(const _Float16* __restrict__ A, int lda,
                               const _Float16* __restrict__ B, int ldb,
                               int m0, int n0, int lane) {
  v8f acc = {};
  #pragma unroll
  for (int k0 = 0; k0 < kDh; k0 += 32) {
    v16h a = load_a_f16(A, lda, m0, k0, lane);
    v16h b = load_b_f16(B, ldb, n0, k0, lane);
    acc = __builtin_amdgcn_wmma_f32_16x16x32_f16(false, a, false, b,
                                                 (short)0, acc, false, false);
  }
  return acc;
}

// P[h][q][e] = sum_d qv[h][q][d] * Wr[e][h*64+d]
__global__ void gemm_p(const _Float16* __restrict__ QV, const _Float16* __restrict__ Wr16,
                       _Float16* __restrict__ P) {
  int h = blockIdx.y;
  int wave = blockIdx.x * 8 + (threadIdx.x >> 5);
  int lane = threadIdx.x & 31;
  int m0 = (wave >> 5) * 16, n0 = (wave & 31) * 16;
  v8f acc = gemm_k64(QV + (size_t)h * kS * kDh, kDh, Wr16 + h * kDh, kD, m0, n0, lane);
  int h4 = lane >> 4, r = lane & 15;
  _Float16* out = P + (size_t)h * kS * kD;
  #pragma unroll
  for (int j = 0; j < 8; j++)
    out[(size_t)(m0 + j + h4 * 8) * kD + n0 + r] = (_Float16)acc[j];
}

// scores[h][q][k] = (q+u)[h][q][:] . K[h][k][:]
__global__ void gemm_ac(const _Float16* __restrict__ QU, const _Float16* __restrict__ KH,
                        float* __restrict__ scores) {
  int h = blockIdx.y;
  int wave = blockIdx.x * 8 + (threadIdx.x >> 5);
  int lane = threadIdx.x & 31;
  int m0 = (wave >> 5) * 16, n0 = (wave & 31) * 16;
  v8f acc = gemm_k64(QU + (size_t)h * kS * kDh, kDh,
                     KH + (size_t)h * kS * kDh, kDh, m0, n0, lane);
  int h4 = lane >> 4, r = lane & 15;
  float* out = scores + (size_t)h * kS * kS;
  #pragma unroll
  for (int j = 0; j < 8; j++)
    out[(size_t)(m0 + j + h4 * 8) * kS + n0 + r] = acc[j];
}

// c[h][q] = qv[h][q][:] . br[h*64:]
__global__ void cbias_kernel(const _Float16* __restrict__ QV,
                             const float* __restrict__ br, float* __restrict__ cb) {
  int idx = blockIdx.x * blockDim.x + threadIdx.x;  // h*kS + q
  if (idx >= kH * kS) return;
  int h = idx >> 9;
  const _Float16* p = QV + (size_t)idx * kDh;
  const float* b = br + h * kDh;
  float s = 0.f;
  #pragma unroll 8
  for (int d = 0; d < kDh; ++d) s += (float)p[d] * b[d];
  cb[idx] = s;
}

// Convert one 16x32 f32 tile row-chunk to an f16 A fragment (cvt in regs).
__device__ inline v16h load_cvt_a(const float* __restrict__ rowbase) {
  v4f f0 = *(const v4f*)rowbase;
  v4f f1 = *(const v4f*)(rowbase + 4);
  v4f f2 = *(const v4f*)(rowbase + 8);
  v4f f3 = *(const v4f*)(rowbase + 12);
  v16h a;
  #pragma unroll
  for (int i = 0; i < 4; i++) {
    a[i]      = (_Float16)f0[i];
    a[i + 4]  = (_Float16)f1[i];
    a[i + 8]  = (_Float16)f2[i];
    a[i + 12] = (_Float16)f3[i];
  }
  return a;
}

// The big streaming pass: scores[h][q][k] += sum_e rel[q][k][e] * P[h][q][e]
// One block per q; rel (536 MB fp32) read exactly once.
// Loop order: e0 outer so the B fragment (P, shared by all 4 m-tiles of the
// wave) is loaded once per e0; 4 accumulators live across the K loop.
__global__ void gemm_bd(const float* __restrict__ rel, const _Float16* __restrict__ P,
                        float* __restrict__ scores) {
  int q = blockIdx.x;
  int wave = threadIdx.x >> 5, lane = threadIdx.x & 31;
  int h4 = lane >> 4, r = lane & 15;
  // wave covers k-rows [wave*64, wave*64+64): 4 tiles of 16
  const float* arow = rel + ((size_t)q * kS + wave * 64 + r) * kD + h4 * 16;
  const _Float16* pb = (r < kH) ? P + ((size_t)r * kS + q) * kD + h4 * 16 : nullptr;
  v8f acc0 = {}, acc1 = {}, acc2 = {}, acc3 = {};
  for (int e0 = 0; e0 < kD; e0 += 32) {
    v16h b = {};
    if (r < kH) b = *(const v16h*)(pb + e0);       // B: column = head (L0-resident)
    __builtin_prefetch(arow + e0 + 128, 0, 0);
    v16h a0 = load_cvt_a(arow + e0);               // mandatory HBM stream
    acc0 = __builtin_amdgcn_wmma_f32_16x16x32_f16(false, a0, false, b,
                                                  (short)0, acc0, false, false);
    v16h a1 = load_cvt_a(arow + 16 * (size_t)kD + e0);
    acc1 = __builtin_amdgcn_wmma_f32_16x16x32_f16(false, a1, false, b,
                                                  (short)0, acc1, false, false);
    v16h a2 = load_cvt_a(arow + 32 * (size_t)kD + e0);
    acc2 = __builtin_amdgcn_wmma_f32_16x16x32_f16(false, a2, false, b,
                                                  (short)0, acc2, false, false);
    v16h a3 = load_cvt_a(arow + 48 * (size_t)kD + e0);
    acc3 = __builtin_amdgcn_wmma_f32_16x16x32_f16(false, a3, false, b,
                                                  (short)0, acc3, false, false);
  }
  if (r < kH) {   // D: lane column = head r, rows m0+h4*8+j (contiguous in k)
    v8f accs[4] = {acc0, acc1, acc2, acc3};
    #pragma unroll
    for (int t = 0; t < 4; t++) {
      float* sp = scores + ((size_t)r * kS + q) * kS + (wave * 4 + t) * 16 + h4 * 8;
      v4f s0 = *(v4f*)sp, s1 = *(v4f*)(sp + 4);
      #pragma unroll
      for (int j = 0; j < 4; j++) { s0[j] += accs[t][j]; s1[j] += accs[t][j + 4]; }
      *(v4f*)sp = s0; *(v4f*)(sp + 4) = s1;
    }
  }
}

// masked softmax over k, +c bias, write f16 attn probs
__global__ void softmax_mask(const float* __restrict__ scores, const float* __restrict__ cbias,
                             const int* __restrict__ rlen, const int* __restrict__ lexn,
                             _Float16* __restrict__ attn) {
  int row = blockIdx.x;                 // h*kS + q
  int tid = threadIdx.x;                // 256 threads, 2 elems each
  int len = rlen[0] + lexn[0];
  float c = cbias[row];
  const float* sp = scores + (size_t)row * kS;
  float x0 = (tid       < len) ? sp[tid]       + c : -1e15f;
  float x1 = (tid + 256 < len) ? sp[tid + 256] + c : -1e15f;
  __shared__ float red[256];
  red[tid] = fmaxf(x0, x1); __syncthreads();
  for (int s = 128; s > 0; s >>= 1) {
    if (tid < s) red[tid] = fmaxf(red[tid], red[tid + s]);
    __syncthreads();
  }
  float mx = red[0]; __syncthreads();
  float e0 = __expf(x0 - mx), e1 = __expf(x1 - mx);
  red[tid] = e0 + e1; __syncthreads();
  for (int s = 128; s > 0; s >>= 1) {
    if (tid < s) red[tid] += red[tid + s];
    __syncthreads();
  }
  float inv = 1.0f / red[0];
  _Float16* ap = attn + (size_t)row * kS;
  ap[tid] = (_Float16)(e0 * inv);
  ap[tid + 256] = (_Float16)(e1 * inv);
}

// out_tmp[q][h*64+d] = sum_k attn[h][q][k] * V[h][k][d]   (B = V^T, K-contiguous)
__global__ void gemm_av(const _Float16* __restrict__ attn, const _Float16* __restrict__ VT,
                        float* __restrict__ outtmp) {
  int h = blockIdx.y;
  int wave = blockIdx.x * 8 + (threadIdx.x >> 5);
  int lane = threadIdx.x & 31;
  int m0 = (wave >> 2) * 16, n0 = (wave & 3) * 16;
  const _Float16* A = attn + (size_t)h * kS * kS;
  const _Float16* B = VT + (size_t)h * kDh * kS;
  v8f acc = {};
  for (int k0 = 0; k0 < kS; k0 += 32) {
    v16h a = load_a_f16(A, kS, m0, k0, lane);
    v16h b = load_b_f16(B, kS, n0, k0, lane);
    acc = __builtin_amdgcn_wmma_f32_16x16x32_f16(false, a, false, b,
                                                 (short)0, acc, false, false);
  }
  int h4 = lane >> 4, r = lane & 15;
  #pragma unroll
  for (int j = 0; j < 8; j++)
    outtmp[(size_t)(m0 + j + h4 * 8) * kD + h * kDh + n0 + r] = acc[j];
}

// out = LayerNorm(residual + attn_out) over D=512 per token
__global__ void layernorm_res(const float* __restrict__ residual,
                              const float* __restrict__ outtmp,
                              float* __restrict__ out) {
  int s = blockIdx.x, tid = threadIdx.x;  // 256 threads, 2 elems each
  const float* rp = residual + (size_t)s * kD;
  const float* op = outtmp + (size_t)s * kD;
  float x0 = rp[tid] + op[tid];
  float x1 = rp[tid + 256] + op[tid + 256];
  __shared__ float red[256];
  red[tid] = x0 + x1; __syncthreads();
  for (int t = 128; t > 0; t >>= 1) {
    if (tid < t) red[tid] += red[tid + t];
    __syncthreads();
  }
  float mu = red[0] * (1.0f / kD); __syncthreads();
  float d0 = x0 - mu, d1 = x1 - mu;
  red[tid] = d0 * d0 + d1 * d1; __syncthreads();
  for (int t = 128; t > 0; t >>= 1) {
    if (tid < t) red[tid] += red[tid + t];
    __syncthreads();
  }
  float rs = rsqrtf(red[0] * (1.0f / kD) + 1e-5f);
  float* o = out + (size_t)s * kD;
  o[tid] = d0 * rs;
  o[tid + 256] = d1 * rs;
}

extern "C" void kernel_launch(void* const* d_in, const int* in_sizes, int n_in,
                              void* d_out, int out_size, void* d_ws, size_t ws_size,
                              hipStream_t stream) {
  const float* query = (const float*)d_in[0];
  const float* key   = (const float*)d_in[1];
  const float* value = (const float*)d_in[2];
  const float* rel   = (const float*)d_in[3];
  const int*   rlen  = (const int*)d_in[4];
  const int*   lexn  = (const int*)d_in[5];
  const float* Wq    = (const float*)d_in[6];
  const float* bq    = (const float*)d_in[7];
  const float* Wk    = (const float*)d_in[8];
  const float* bk    = (const float*)d_in[9];
  const float* Wv    = (const float*)d_in[10];
  const float* bv    = (const float*)d_in[11];
  const float* Wr    = (const float*)d_in[12];
  const float* br    = (const float*)d_in[13];
  const float* u     = (const float*)d_in[14];
  const float* v     = (const float*)d_in[15];
  float* out = (float*)d_out;

  char* ws = (char*)d_ws;
  _Float16* XQ  = (_Float16*)(ws + OFF_XQ);
  _Float16* XK  = (_Float16*)(ws + OFF_XK);
  _Float16* XV  = (_Float16*)(ws + OFF_XV);
  _Float16* WQT = (_Float16*)(ws + OFF_WQT);
  _Float16* WKT = (_Float16*)(ws + OFF_WKT);
  _Float16* WVT = (_Float16*)(ws + OFF_WVT);
  _Float16* WR  = (_Float16*)(ws + OFF_WR);
  _Float16* QU  = (_Float16*)(ws + OFF_QU);
  _Float16* QV  = (_Float16*)(ws + OFF_QV);
  _Float16* KH  = (_Float16*)(ws + OFF_KH);
  _Float16* VT  = (_Float16*)(ws + OFF_VT);
  _Float16* P   = (_Float16*)(ws + OFF_P);
  float*    SC  = (float*)(ws + OFF_SC);
  _Float16* AT  = (_Float16*)(ws + OFF_AT);
  float*    CB  = (float*)(ws + OFF_CB);
  float*    OT  = (float*)(ws + OFF_OT);

  const int nElem = kS * kD;            // 262144
  // 1) convert activations + Wr to f16
  cvt_f32_to_f16<<<nElem / 256, 256, 0, stream>>>(query, XQ, nElem);
  cvt_f32_to_f16<<<nElem / 256, 256, 0, stream>>>(key,   XK, nElem);
  cvt_f32_to_f16<<<nElem / 256, 256, 0, stream>>>(value, XV, nElem);
  cvt_f32_to_f16<<<nElem / 256, 256, 0, stream>>>(Wr,    WR, nElem);
  // 2) transpose+convert projection weights
  dim3 tg(16, 16);
  transpose_cvt<<<tg, 256, 0, stream>>>(Wq, WQT);
  transpose_cvt<<<tg, 256, 0, stream>>>(Wk, WKT);
  transpose_cvt<<<tg, 256, 0, stream>>>(Wv, WVT);
  // 3) QKV projections (WMMA), fusing +u/+v and V-transpose
  gemm_proj<<<128, 256, 0, stream>>>(XQ, WQT, bq, u, v, QU, QV, 0);
  gemm_proj<<<128, 256, 0, stream>>>(XK, WKT, bk, nullptr, nullptr, KH, nullptr, 1);
  gemm_proj<<<128, 256, 0, stream>>>(XV, WVT, bv, nullptr, nullptr, VT, nullptr, 2);
  // 4) P[h][q][e] and bias c[h][q]
  gemm_p<<<dim3(128, kH), 256, 0, stream>>>(QV, WR, P);
  cbias_kernel<<<(kH * kS) / 256, 256, 0, stream>>>(QV, br, CB);
  // 5) A_C scores
  gemm_ac<<<dim3(128, kH), 256, 0, stream>>>(QU, KH, SC);
  // 6) B_D: single streaming pass over rel (the only heavy data movement)
  gemm_bd<<<kS, 256, 0, stream>>>(rel, P, SC);
  // 7) masked softmax
  softmax_mask<<<kH * kS, 256, 0, stream>>>(SC, CB, rlen, lexn, AT);
  // 8) attn @ V
  gemm_av<<<dim3(16, kH), 256, 0, stream>>>(AT, VT, OT);
  // 9) residual + LayerNorm
  layernorm_res<<<kS, 256, 0, stream>>>(query, OT, out);
  (void)in_sizes; (void)n_in; (void)out_size; (void)ws_size;
}
